// Model_39676907884756
// MI455X (gfx1250) — matrix operand order repaired and checked by reading, not compiled, not run on hardware
//
#include <hip/hip_runtime.h>
#include <hip/hip_bf16.h>
#include <math.h>
#include <stdint.h>

typedef __attribute__((ext_vector_type(16))) __bf16 v16bf;
typedef __attribute__((ext_vector_type(8)))  __bf16 v8bf;
typedef __attribute__((ext_vector_type(4)))  __bf16 v4bf;
typedef __attribute__((ext_vector_type(8)))  float  v8f;
typedef __attribute__((ext_vector_type(4)))  float  v4f;

#define D_DIM 1024
#define S_DIM 2048
#define B_DIM 8

// GEMM tiling
#define BM 128
#define BN 64
#define BK 32
#define LDK 40   // padded LDS row (bf16 elems): 80B rows keep every 16B chunk aligned

__device__ __forceinline__ v16bf cat8(v8bf lo, v8bf hi) {
    return __builtin_shufflevector(lo, hi, 0,1,2,3,4,5,6,7,8,9,10,11,12,13,14,15);
}

__device__ __forceinline__ v8f wmma_bf16(v16bf a, v16bf b, v8f c) {
    return __builtin_amdgcn_wmma_f32_16x16x32_bf16(false, a, false, b, (short)0, c, false, false);
}

// CDNA5 async global->LDS copy, 16 bytes per lane (tracked by ASYNCcnt).
__device__ __forceinline__ void async_ld16(uint32_t lds_off, const void* g) {
    unsigned long long ga = (unsigned long long)g;
    asm volatile("global_load_async_to_lds_b128 %0, %1, off"
                 :: "v"(lds_off), "v"(ga) : "memory");
}

__device__ __forceinline__ void wait_async0() {
    asm volatile("s_wait_asynccnt 0x0" ::: "memory");
}

__device__ __forceinline__ uint32_t lds_off_of(const void* p) {
    // generic LDS pointer = {shared aperture, lds offset} -> low 32 bits
    return (uint32_t)(size_t)p;
}

// ---------------------------------------------------------------------------
// fp32 -> bf16 elementwise (n multiple of 2048)
// ---------------------------------------------------------------------------
__global__ __launch_bounds__(256)
void cvt_f32_bf16(const float* __restrict__ in, __bf16* __restrict__ out, size_t n)
{
    size_t i = ((size_t)blockIdx.x * 256 + threadIdx.x) * 8;
    if (i + 8 <= n) {
        v4f a = *(const v4f*)(in + i);
        v4f b = *(const v4f*)(in + i + 4);
        v8bf o;
        o[0] = (__bf16)a[0]; o[1] = (__bf16)a[1]; o[2] = (__bf16)a[2]; o[3] = (__bf16)a[3];
        o[4] = (__bf16)b[0]; o[5] = (__bf16)b[1]; o[6] = (__bf16)b[2]; o[7] = (__bf16)b[3];
        *(v8bf*)(out + i) = o;
    }
}

// ---------------------------------------------------------------------------
// vT[b][d][key] (bf16) = v1[b][key][d] (fp32); 32x32 tiles via LDS
// ---------------------------------------------------------------------------
__global__ __launch_bounds__(256)
void transpose_v_bf16(const float* __restrict__ V, __bf16* __restrict__ VT)
{
    __shared__ __bf16 t[32][33];
    const int b  = blockIdx.z;
    const int k0 = blockIdx.x * 32;   // key
    const int d0 = blockIdx.y * 32;   // d
    const int tid = threadIdx.x;

    {
        int key = tid >> 3;
        int c4  = (tid & 7) * 4;
        v4f f = *(const v4f*)(V + ((size_t)b * S_DIM + k0 + key) * D_DIM + d0 + c4);
        t[c4 + 0][key] = (__bf16)f[0];
        t[c4 + 1][key] = (__bf16)f[1];
        t[c4 + 2][key] = (__bf16)f[2];
        t[c4 + 3][key] = (__bf16)f[3];
    }
    __syncthreads();
    {
        int d  = tid >> 3;
        int kc = (tid & 7) * 4;
        v4bf o;
        o[0] = t[d][kc + 0]; o[1] = t[d][kc + 1];
        o[2] = t[d][kc + 2]; o[3] = t[d][kc + 3];
        *(v4bf*)(VT + ((size_t)b * D_DIM + d0 + d) * S_DIM + k0 + kc) = o;
    }
}

// ---------------------------------------------------------------------------
// Unified bf16 GEMM: C[m,n] = sum_k A[m,k] * Bw[n,k] (+ bias[n])
// Double-buffered LDS staging via GLOBAL_LOAD_ASYNC_TO_LDS_B128.
// Block tile 128x64, 8 waves of 32x32, K stepped by 32. Batched via blockIdx.z.
// ---------------------------------------------------------------------------
template<bool OUT_F32>
__global__ __launch_bounds__(256)
void gemm_bf16(const __bf16* __restrict__ A, const __bf16* __restrict__ Bw,
               const float* __restrict__ bias, void* __restrict__ Cv,
               int M, int N, int K,
               size_t batchA, size_t batchB, size_t batchC)
{
    __shared__ __bf16 ldsA[2][BM][LDK];
    __shared__ __bf16 ldsB[2][BN][LDK];

    const int tid   = threadIdx.x;
    const int lane  = tid & 31;
    const int wave  = tid >> 5;
    const int wm    = wave >> 1;      // 0..3
    const int wn    = wave & 1;       // 0..1
    const int l16   = lane & 15;
    const int khalf = lane >> 4;
    const int m0    = blockIdx.y * BM;
    const int n0    = blockIdx.x * BN;

    const __bf16* Ab = A  + (size_t)blockIdx.z * batchA;
    const __bf16* Bb = Bw + (size_t)blockIdx.z * batchB;

    // per-thread staging coordinates (16B chunks; 4 chunks per 32-elem row)
    const int ar0 = tid >> 2;               // A rows: tid/4 and +64
    const int ac  = (tid & 3) * 8;
    const int br0 = tid >> 2;               // B rows: 0..63
    const int bc  = (tid & 3) * 8;

    v8f acc[2][2] = {};

    const int nk = K / BK;

    // prologue: stage tile 0 into buffer 0
    {
        async_ld16(lds_off_of(&ldsA[0][ar0][ac]),      Ab + (size_t)(m0 + ar0) * K + ac);
        async_ld16(lds_off_of(&ldsA[0][ar0 + 64][ac]), Ab + (size_t)(m0 + ar0 + 64) * K + ac);
        async_ld16(lds_off_of(&ldsB[0][br0][bc]),      Bb + (size_t)(n0 + br0) * K + bc);
    }

    for (int i = 0; i < nk; ++i) {
        wait_async0();
        __syncthreads();

        const int cur = i & 1;
        if (i + 1 < nk) {
            const int nxt = cur ^ 1;
            const int k1  = (i + 1) * BK;
            async_ld16(lds_off_of(&ldsA[nxt][ar0][ac]),      Ab + (size_t)(m0 + ar0) * K + k1 + ac);
            async_ld16(lds_off_of(&ldsA[nxt][ar0 + 64][ac]), Ab + (size_t)(m0 + ar0 + 64) * K + k1 + ac);
            async_ld16(lds_off_of(&ldsB[nxt][br0][bc]),      Bb + (size_t)(n0 + br0) * K + k1 + bc);
        }

        v16bf afrag[2], bfrag[2];
        #pragma unroll
        for (int a = 0; a < 2; ++a) {
            const __bf16* ab = &ldsA[cur][wm * 32 + a * 16 + l16][khalf * 8];
            afrag[a] = cat8(*(const v8bf*)ab, *(const v8bf*)(ab + 16));
        }
        #pragma unroll
        for (int j = 0; j < 2; ++j) {
            const __bf16* bb = &ldsB[cur][wn * 32 + j * 16 + l16][khalf * 8];
            bfrag[j] = cat8(*(const v8bf*)bb, *(const v8bf*)(bb + 16));
        }
        #pragma unroll
        for (int a = 0; a < 2; ++a)
            #pragma unroll
            for (int j = 0; j < 2; ++j)
                acc[a][j] = wmma_bf16(afrag[a], bfrag[j], acc[a][j]);
        // no trailing barrier: next iteration's wait+barrier protects both buffers
    }

    #pragma unroll
    for (int a = 0; a < 2; ++a) {
        #pragma unroll
        for (int j = 0; j < 2; ++j) {
            int cc = n0 + wn * 32 + j * 16 + l16;
            float bv = bias ? bias[cc] : 0.0f;
            #pragma unroll
            for (int e = 0; e < 8; ++e) {
                int rr = m0 + wm * 32 + a * 16 + khalf * 8 + e;
                float v = acc[a][j][e] + bv;
                if (OUT_F32)
                    ((float*)Cv)[(size_t)blockIdx.z * batchC + (size_t)rr * N + cc] = v;
                else
                    ((__bf16*)Cv)[(size_t)blockIdx.z * batchC + (size_t)rr * N + cc] = (__bf16)v;
            }
        }
    }
}

// ---------------------------------------------------------------------------
// Scores + softmax: block owns 16 query rows of one batch; wave w owns keys
// [w*256, (w+1)*256). P[b,q,key] = softmax_row(qp.kp^T/sqrt(D) + mask), bf16.
// ---------------------------------------------------------------------------
__global__ __launch_bounds__(256)
void attn_scores_softmax(const __bf16* __restrict__ Qp,
                         const __bf16* __restrict__ Kp,
                         const float*  __restrict__ mask,
                         __bf16* __restrict__ P)
{
    __shared__ __bf16 qtile[16][D_DIM];   // 32 KB
    __shared__ float  red[16][8];

    const int b     = blockIdx.y;
    const int q0    = blockIdx.x * 16;
    const int tid   = threadIdx.x;
    const int lane  = tid & 31;
    const int wave  = tid >> 5;
    const int l16   = lane & 15;
    const int khalf = lane >> 4;

    // async-stage 16 query rows (bf16) into LDS
    const __bf16* Qb = Qp + ((size_t)b * S_DIM + q0) * D_DIM;
    #pragma unroll
    for (int p = 0; p < 8; ++p) {
        int idx = tid + p * 256;            // 2048 chunks of 16B
        int r = idx >> 7;
        int c = (idx & 127) * 8;
        async_ld16(lds_off_of(&qtile[r][c]), Qb + (size_t)r * D_DIM + c);
    }
    wait_async0();
    __syncthreads();

    const int keybase = wave * 256;
    const __bf16* Kb = Kp + (size_t)b * S_DIM * D_DIM;

    v8f acc[16] = {};
    for (int d0 = 0; d0 < D_DIM; d0 += 32) {
        const __bf16* ab = &qtile[l16][d0 + khalf * 8];
        v16bf afrag = cat8(*(const v8bf*)ab, *(const v8bf*)(ab + 16));
        #pragma unroll
        for (int t = 0; t < 16; ++t) {
            const __bf16* bb =
                Kb + (size_t)(keybase + t * 16 + l16) * D_DIM + d0 + khalf * 8;
            v16bf bfrag = cat8(*(const v8bf*)bb, *(const v8bf*)(bb + 16));
            acc[t] = wmma_bf16(afrag, bfrag, acc[t]);
        }
    }

    // scale + mask (mask is [1,S,S], broadcast over batch)
    const float scale = 0.03125f;  // 1/sqrt(1024)
    #pragma unroll
    for (int t = 0; t < 16; ++t) {
        #pragma unroll
        for (int e = 0; e < 8; ++e) {
            int qr  = q0 + khalf * 8 + e;
            int key = keybase + t * 16 + l16;
            acc[t][e] = acc[t][e] * scale + mask[(size_t)qr * S_DIM + key];
        }
    }

    // row max: lane-local over tiles -> shuffle-xor over 16-lane half -> LDS over waves
    float m[8];
    #pragma unroll
    for (int e = 0; e < 8; ++e) {
        float v = -__builtin_inff();
        #pragma unroll
        for (int t = 0; t < 16; ++t) v = fmaxf(v, acc[t][e]);
        m[e] = v;
    }
    #pragma unroll
    for (int s = 1; s <= 8; s <<= 1)
        #pragma unroll
        for (int e = 0; e < 8; ++e)
            m[e] = fmaxf(m[e], __shfl_xor(m[e], s, 32));
    if (l16 == 0) {
        #pragma unroll
        for (int e = 0; e < 8; ++e) red[khalf * 8 + e][wave] = m[e];
    }
    __syncthreads();
    float M[8];
    #pragma unroll
    for (int e = 0; e < 8; ++e) {
        float v = -__builtin_inff();
        #pragma unroll
        for (int w = 0; w < 8; ++w) v = fmaxf(v, red[khalf * 8 + e][w]);
        M[e] = v;
    }
    __syncthreads();

    // exp + row sum
    float sum[8] = {};
    #pragma unroll
    for (int t = 0; t < 16; ++t) {
        #pragma unroll
        for (int e = 0; e < 8; ++e) {
            float p = __expf(acc[t][e] - M[e]);
            acc[t][e] = p;
            sum[e] += p;
        }
    }
    #pragma unroll
    for (int s = 1; s <= 8; s <<= 1)
        #pragma unroll
        for (int e = 0; e < 8; ++e)
            sum[e] += __shfl_xor(sum[e], s, 32);
    if (l16 == 0) {
        #pragma unroll
        for (int e = 0; e < 8; ++e) red[khalf * 8 + e][wave] = sum[e];
    }
    __syncthreads();
    float rinv[8];
    #pragma unroll
    for (int e = 0; e < 8; ++e) {
        float v = 0.f;
        #pragma unroll
        for (int w = 0; w < 8; ++w) v += red[khalf * 8 + e][w];
        rinv[e] = 1.0f / v;
    }

    // store normalized probs as bf16
    #pragma unroll
    for (int t = 0; t < 16; ++t) {
        #pragma unroll
        for (int e = 0; e < 8; ++e) {
            int qr  = q0 + khalf * 8 + e;
            int key = keybase + t * 16 + l16;
            P[((size_t)b * S_DIM + qr) * S_DIM + key] = (__bf16)(acc[t][e] * rinv[e]);
        }
    }
}

// ---------------------------------------------------------------------------
extern "C" void kernel_launch(void* const* d_in, const int* in_sizes, int n_in,
                              void* d_out, int out_size, void* d_ws, size_t ws_size,
                              hipStream_t stream)
{
    const float* q    = (const float*)d_in[0];
    const float* k    = (const float*)d_in[1];
    const float* v1   = (const float*)d_in[2];
    const float* mask = (const float*)d_in[3];
    const float* W1   = (const float*)d_in[4];
    const float* b1   = (const float*)d_in[5];
    const float* W2   = (const float*)d_in[6];
    const float* b2   = (const float*)d_in[7];
    float* out = (float*)d_out;

    char* ws = (char*)d_ws;
    const size_t MTOK = (size_t)B_DIM * S_DIM;           // 16384 tokens
    const size_t ACT  = MTOK * D_DIM * sizeof(__bf16);   // 32 MB
    const size_t WB   = (size_t)D_DIM * D_DIM * sizeof(__bf16); // 2 MB

    __bf16* qb  = (__bf16*)(ws);
    __bf16* kb  = (__bf16*)(ws + 1 * ACT);
    __bf16* vT  = (__bf16*)(ws + 2 * ACT);
    __bf16* w1b = (__bf16*)(ws + 3 * ACT);
    __bf16* w2b = (__bf16*)(ws + 3 * ACT + WB);
    __bf16* hq  = (__bf16*)(ws + 3 * ACT + 2 * WB);
    __bf16* hk  = (__bf16*)(ws + 4 * ACT + 2 * WB);
    __bf16* qp  = (__bf16*)(ws + 5 * ACT + 2 * WB);
    __bf16* kp  = (__bf16*)(ws + 6 * ACT + 2 * WB);
    // P (B*S*S bf16 = 64 MB) reuses qb+kb, dead by the time it is written.
    __bf16* P   = (__bf16*)ws;

    dim3 blk(256);

    // ---- fp32 -> bf16 conversions / v1 transpose ----
    {
        size_t nAct = MTOK * D_DIM;           // 16,777,216
        size_t nW   = (size_t)D_DIM * D_DIM;  //  1,048,576
        cvt_f32_bf16<<<dim3((unsigned)(nAct / 2048)), blk, 0, stream>>>(q,  qb,  nAct);
        cvt_f32_bf16<<<dim3((unsigned)(nAct / 2048)), blk, 0, stream>>>(k,  kb,  nAct);
        cvt_f32_bf16<<<dim3((unsigned)(nW   / 2048)), blk, 0, stream>>>(W1, w1b, nW);
        cvt_f32_bf16<<<dim3((unsigned)(nW   / 2048)), blk, 0, stream>>>(W2, w2b, nW);
        transpose_v_bf16<<<dim3(S_DIM / 32, D_DIM / 32, B_DIM), blk, 0, stream>>>(v1, vT);
    }

    // ---- projections: h = x@W1^T+b1 ; p = h@W2^T+b2 (all bf16, async staged) ----
    dim3 gproj(D_DIM / BN, (unsigned)(MTOK / BM), 1);     // (16, 128, 1)
    gemm_bf16<false><<<gproj, blk, 0, stream>>>(qb, w1b, b1, hq, (int)MTOK, D_DIM, D_DIM, 0, 0, 0);
    gemm_bf16<false><<<gproj, blk, 0, stream>>>(kb, w1b, b1, hk, (int)MTOK, D_DIM, D_DIM, 0, 0, 0);
    gemm_bf16<false><<<gproj, blk, 0, stream>>>(hq, w2b, b2, qp, (int)MTOK, D_DIM, D_DIM, 0, 0, 0);
    gemm_bf16<false><<<gproj, blk, 0, stream>>>(hk, w2b, b2, kp, (int)MTOK, D_DIM, D_DIM, 0, 0, 0);

    // ---- scores + softmax -> P (bf16) ----
    dim3 gattn(S_DIM / 16, B_DIM);                        // (128, 8)
    attn_scores_softmax<<<gattn, blk, 0, stream>>>(qp, kp, mask, P);

    // ---- out = P @ v1 via vT (batched) ----
    dim3 gpv(D_DIM / BN, S_DIM / BM, B_DIM);              // (16, 16, 8)
    gemm_bf16<true><<<gpv, blk, 0, stream>>>(
        P, vT, nullptr, out, S_DIM, D_DIM, S_DIM,
        (size_t)S_DIM * S_DIM, (size_t)D_DIM * S_DIM, (size_t)S_DIM * D_DIM);
}
